// CrossModalMamba_4595615007329
// MI455X (gfx1250) — matrix operand order repaired and checked by reading, not compiled, not run on hardware
//
#include <hip/hip_runtime.h>
#include <hip/hip_bf16.h>
#include <math.h>

// ---------------- model constants ----------------
#define BN   128   // batch
#define CC   128   // channels
#define LL   81    // spatial tokens (9*9)
#define DSN  32    // mamba state size
#define DCN  4     // conv width
#define HEADS 4
#define DH   32    // head dim
#define T1   (BN*LL)   // 10368 tokens (multiple of 16)
#define T2   (BN*CC)   // 16384 tokens (multiple of 16)
#define DI1  256
#define DI2  162
#define R1n  8
#define R2n  6
#define W1n  (R1n + 2*DSN)  // 72
#define W2n  (R2n + 2*DSN)  // 70

typedef __attribute__((ext_vector_type(16))) _Float16 v16h;
typedef __attribute__((ext_vector_type(8)))  float    v8f;
typedef __attribute__((ext_vector_type(4)))  float    f32x4;

// ---------------------------------------------------------------------
// gfx1250 async global->LDS copy (ASYNCcnt-tracked), §15.18 opcode 96.
// LDS generic pointers carry the LDS byte offset in their low 32 bits.
// ---------------------------------------------------------------------
__device__ __forceinline__ void async_g2l_b32(void* ldsp, const void* gp)
{
  unsigned int loff = (unsigned int)(unsigned long long)ldsp;
  unsigned long long ga = (unsigned long long)gp;
  asm volatile("global_load_async_to_lds_b32 %0, %1, off"
               :: "v"(loff), "v"(ga) : "memory");
}
__device__ __forceinline__ void wait_async0()
{
  asm volatile("s_wait_asynccnt 0x0" ::: "memory");
}

// =====================================================================
// Generic WMMA GEMM:  Out[M,N] = A[M,K] @ Wt[N,K]^T  (+bias, +softplus)
// One wave per 16x16 tile; fp32 memory, f16 MACs, fp32 accum via
// v_wmma_f32_16x16x32_f16.  Caller guarantees M % 16 == 0.
// AL16: base pointers 16B-aligned AND lda/ldw % 4 == 0 -> float4 loads.
// ep: 0 = none, 1 = +bias, 2 = softplus(x + bias)
// =====================================================================
template <bool AL16>
__global__ __launch_bounds__(32) void k_gemm_wmma(
    const float* __restrict__ A, int lda,
    const float* __restrict__ Wt, int ldw,
    const float* __restrict__ bias,
    float* __restrict__ Out, int ldo,
    int M, int N, int K, int ep)
{
  const int lane = threadIdx.x & 31;
  const int idx  = lane & 15;
  const int g    = lane >> 4;
  const int m0 = blockIdx.y * 16, n0 = blockIdx.x * 16;
  const int mA = m0 + idx;                  // always < M (M % 16 == 0)
  const int nB = n0 + idx;
  const bool nOK = (nB < N);

  const float* Arow = A  + (size_t)mA * lda;
  const float* Brow = Wt + (size_t)(nOK ? nB : (N - 1)) * ldw;  // clamped, loads stay legal

  v8f acc = {};
  const int Kfull = K & ~31;

  for (int kb = 0; kb < Kfull; kb += 32) {
    __builtin_prefetch(Arow + kb + 32, 0, 0);   // gfx1250 global_prefetch_b8
    v16h a, b;
    if (AL16) {
      // A lane data: 16 contiguous floats at [kb+16g): 4 aligned float4 loads
      const f32x4* ap = (const f32x4*)(Arow + kb);
      f32x4 a0 = ap[4 * g + 0];
      f32x4 a1 = ap[4 * g + 1];
      f32x4 a2 = ap[4 * g + 2];
      f32x4 a3 = ap[4 * g + 3];
      // B lane data: 16 contiguous floats at [kb+16g): 4 aligned float4 loads
      const f32x4* bp = (const f32x4*)(Brow + kb + 16 * g);
      f32x4 b0 = bp[0], b1 = bp[1], b2 = bp[2], b3 = bp[3];
#pragma unroll
      for (int j = 0; j < 4; ++j) {
        a[j]      = (_Float16)a0[j];
        a[4 + j]  = (_Float16)a1[j];
        a[8 + j]  = (_Float16)a2[j];
        a[12 + j] = (_Float16)a3[j];
        b[j]      = (_Float16)b0[j];
        b[4 + j]  = (_Float16)b1[j];
        b[8 + j]  = (_Float16)b2[j];
        b[12 + j] = (_Float16)b3[j];
      }
    } else {
#pragma unroll
      for (int h = 0; h < 16; ++h) {
        // A operand, this code's k order: k = h + 16g
        int ka = kb + h + 16 * g;
        a[h] = (_Float16)Arow[ka];
        // B operand, same k order as A: lanes 0-15 k = 0..15, lanes 16-31 k = 16..31
        int kw = kb + h + 16 * g;
        b[h] = (_Float16)Brow[kw];
      }
    }
    acc = __builtin_amdgcn_wmma_f32_16x16x32_f16(false, a, false, b,
                                                 (short)0, acc, false, false);
  }

  if (K & 31) {     // branch-free tail: clamp address, select value
    v16h a, b;
#pragma unroll
    for (int h = 0; h < 16; ++h) {
      int ka = Kfull + h + 16 * g;
      int kac = (ka < K) ? ka : (K - 1);
      float av = Arow[kac];
      a[h] = (_Float16)((ka < K) ? av : 0.0f);
      int kw = Kfull + h + 16 * g;
      int kwc = (kw < K) ? kw : (K - 1);
      float bv = Brow[kwc];
      b[h] = (_Float16)((kw < K) ? bv : 0.0f);
    }
    acc = __builtin_amdgcn_wmma_f32_16x16x32_f16(false, a, false, b,
                                                 (short)0, acc, false, false);
  }

#pragma unroll
  for (int r = 0; r < 8; ++r) {
    int m = m0 + r + 8 * g;
    int n = n0 + idx;
    if (n < N) {
      float v = acc[r];
      if (ep >= 1 && bias) v += bias[n];
      if (ep == 2) v = (v > 20.0f) ? v : log1pf(__expf(v));
      Out[(size_t)m * ldo + n] = v;
    }
  }
}

// =====================================================================
// seq[b,l,c] = LayerNorm_c( x[b,c,l] )   (fused transpose + ln_pre)
// =====================================================================
__global__ __launch_bounds__(128) void k_seq_ln(
    const float* __restrict__ x, const float* __restrict__ gam,
    const float* __restrict__ bet, float* __restrict__ seq)
{
  int row = blockIdx.x;           // b*LL + l
  int l = row % LL, b = row / LL;
  int c = threadIdx.x;
  __shared__ float red[128];
  float v = x[((size_t)b * CC + c) * LL + l];
  red[c] = v; __syncthreads();
  for (int s = 64; s > 0; s >>= 1) { if (c < s) red[c] += red[c + s]; __syncthreads(); }
  float mu = red[0] / (float)CC; __syncthreads();
  float xc = v - mu;
  red[c] = xc * xc; __syncthreads();
  for (int s = 64; s > 0; s >>= 1) { if (c < s) red[c] += red[c + s]; __syncthreads(); }
  float var = red[0] / (float)CC;
  seq[(size_t)row * CC + c] = xc * rsqrtf(var + 1e-5f) * gam[c] + bet[c];
}

// =====================================================================
// out = LayerNorm_c(a [+ b2] [+ c3]) over width 128
// =====================================================================
__global__ __launch_bounds__(128) void k_ln3(
    const float* __restrict__ a, const float* __restrict__ b2,
    const float* __restrict__ c3, const float* __restrict__ gam,
    const float* __restrict__ bet, float* __restrict__ out)
{
  int row = blockIdx.x;
  int c = threadIdx.x;
  size_t off = (size_t)row * CC + c;
  __shared__ float red[128];
  float v = a[off];
  if (b2) v += b2[off];
  if (c3) v += c3[off];
  red[c] = v; __syncthreads();
  for (int s = 64; s > 0; s >>= 1) { if (c < s) red[c] += red[c + s]; __syncthreads(); }
  float mu = red[0] / (float)CC; __syncthreads();
  float xc = v - mu;
  red[c] = xc * xc; __syncthreads();
  for (int s = 64; s > 0; s >>= 1) { if (c < s) red[c] += red[c + s]; __syncthreads(); }
  float var = red[0] / (float)CC;
  out[off] = xc * rsqrtf(var + 1e-5f) * gam[c] + bet[c];
}

// =====================================================================
// causal depthwise conv (width 4) + SiLU
// =====================================================================
__global__ void k_conv_silu(const float* __restrict__ xz,
                            const float* __restrict__ cw,
                            const float* __restrict__ cb,
                            float* __restrict__ u,
                            int Bn, int T, int di)
{
  size_t i = (size_t)blockIdx.x * blockDim.x + threadIdx.x;
  size_t tot = (size_t)Bn * T * di;
  if (i >= tot) return;
  int ch = (int)(i % di);
  size_t bt = i / di;
  int t = (int)(bt % T);
  int b = (int)(bt / T);
  float acc = cb[ch];
#pragma unroll
  for (int k = 0; k < DCN; ++k) {
    int tt = t + k - (DCN - 1);
    if (tt >= 0)
      acc += xz[((size_t)b * T + tt) * (size_t)(2 * di) + ch] * cw[ch * DCN + k];
  }
  u[i] = acc / (1.0f + __expf(-acc));   // SiLU
}

// =====================================================================
// selective scan, fused (y + u*D)*silu(z) epilogue
// =====================================================================
__global__ void k_scan(const float* __restrict__ u,  const float* __restrict__ dt,
                       const float* __restrict__ xdbl, int Wd, int r,
                       const float* __restrict__ A_log, const float* __restrict__ Dp,
                       const float* __restrict__ xz,
                       float* __restrict__ ym, int T, int di)
{
  int b = blockIdx.x;
  int d = threadIdx.x;
  bool act = d < di;
  __shared__ float Bs[DSN], Cs[DSN];
  float Ac[DSN], h[DSN];
  float Dv = 0.0f;
  if (act) {
#pragma unroll
    for (int s = 0; s < DSN; ++s) { Ac[s] = -__expf(A_log[d * DSN + s]); h[s] = 0.0f; }
    Dv = Dp[d];
  }
  for (int t = 0; t < T; ++t) {
    __syncthreads();
    if (threadIdx.x < DSN) {
      size_t tok = (size_t)b * T + t;
      Bs[threadIdx.x] = xdbl[tok * Wd + r + threadIdx.x];
      Cs[threadIdx.x] = xdbl[tok * Wd + r + DSN + threadIdx.x];
    }
    __syncthreads();
    if (act) {
      size_t tok = (size_t)b * T + t;
      float dtv = dt[tok * di + d];
      float uv  = u[tok * di + d];
      float du  = dtv * uv;
      float y = 0.0f;
#pragma unroll
      for (int s = 0; s < DSN; ++s) {
        h[s] = __expf(dtv * Ac[s]) * h[s] + du * Bs[s];
        y += h[s] * Cs[s];
      }
      float zv = xz[tok * (size_t)(2 * di) + di + d];
      float sz = zv / (1.0f + __expf(-zv));
      ym[tok * di + d] = (y + uv * Dv) * sz;
    }
  }
}

// =====================================================================
// transposes between (B,L,C) and (B,C,L)
// =====================================================================
__global__ void k_t_blc2bcl(const float* __restrict__ in, float* __restrict__ out)
{
  size_t i = (size_t)blockIdx.x * blockDim.x + threadIdx.x;
  if (i >= (size_t)BN * LL * CC) return;
  int c = (int)(i % CC); size_t bl = i / CC;
  int l = (int)(bl % LL); int b = (int)(bl / LL);
  out[((size_t)b * CC + c) * LL + l] = in[i];
}
__global__ void k_t_bcl2blc(const float* __restrict__ in, float* __restrict__ out)
{
  size_t i = (size_t)blockIdx.x * blockDim.x + threadIdx.x;
  if (i >= (size_t)BN * CC * LL) return;
  int l = (int)(i % LL); size_t bc = i / LL;
  int c = (int)(bc % CC); int b = (int)(bc / CC);
  out[((size_t)b * LL + l) * CC + c] = in[i];
}

// =====================================================================
// fused batch-axis attention, one block per (token l, head h, q-half).
// Q/K/V staged to LDS via gfx1250 async global->LDS copies, then f16
// fragments feed v_wmma_f32_16x16x32_f16 for S = QK^T and O = PV.
// =====================================================================
__global__ __launch_bounds__(256) void k_attn(
    const float* __restrict__ Q, const float* __restrict__ Kk,
    const float* __restrict__ V, float* __restrict__ O)
{
  int lh = blockIdx.x;
  int l = lh / HEADS, h = lh % HEADS;
  int qc = blockIdx.y;                     // q rows [qc*64, qc*64+64)
  __shared__ _Float16 Qh[64 * DH];
  __shared__ _Float16 Kh[128 * DH];
  __shared__ _Float16 Vh[128 * DH];
  __shared__ float    Sf[64 * 128];        // scores; doubles as f32 staging
  int tid = threadIdx.x;

  // ---- phase A: async-stage K and V (4096 f32 each) into Sf ----
  for (int i = tid; i < 128 * DH; i += 256) {
    int kb = i >> 5, d = i & 31;
    size_t off = ((size_t)kb * LL + l) * CC + h * DH + d;
    async_g2l_b32(&Sf[i],        Kk + off);
    async_g2l_b32(&Sf[4096 + i], V  + off);
  }
  wait_async0();
  __syncthreads();
  for (int i = tid; i < 128 * DH; i += 256) {
    Kh[i] = (_Float16)Sf[i];
    Vh[i] = (_Float16)Sf[4096 + i];
  }
  __syncthreads();

  // ---- phase B: async-stage Q (2048 f32) ----
  for (int i = tid; i < 64 * DH; i += 256) {
    int qb = qc * 64 + (i >> 5); int d = i & 31;
    async_g2l_b32(&Sf[i], Q + ((size_t)qb * LL + l) * CC + h * DH + d);
  }
  wait_async0();
  __syncthreads();
  for (int i = tid; i < 64 * DH; i += 256) Qh[i] = (_Float16)Sf[i];
  __syncthreads();

  int wave = tid >> 5, lane = tid & 31, idx = lane & 15, g = lane >> 4;
  const float scale = 0.17677669529663687f;   // 1/sqrt(32)

  // S = Q K^T : 64x128 = 4x8 tiles; wave w does 4 tiles (K=32 -> 1 WMMA each)
  for (int t = wave * 4; t < wave * 4 + 4; ++t) {
    int tm = t >> 3, tn = t & 7;
    v16h a, b; v8f acc = {};
#pragma unroll
    for (int hh = 0; hh < 16; ++hh) {
      int ka = hh + 16 * g;
      a[hh] = Qh[(tm * 16 + idx) * DH + ka];
      int kw = hh + 16 * g;
      b[hh] = Kh[(tn * 16 + idx) * DH + kw];
    }
    acc = __builtin_amdgcn_wmma_f32_16x16x32_f16(false, a, false, b,
                                                 (short)0, acc, false, false);
#pragma unroll
    for (int r = 0; r < 8; ++r)
      Sf[(tm * 16 + r + 8 * g) * 128 + tn * 16 + idx] = acc[r] * scale;
  }
  __syncthreads();

  // row softmax (64 rows, one thread each)
  if (tid < 64) {
    float mx = -1e30f;
    for (int j = 0; j < 128; ++j) mx = fmaxf(mx, Sf[tid * 128 + j]);
    float sum = 0.0f;
    for (int j = 0; j < 128; ++j) {
      float e = __expf(Sf[tid * 128 + j] - mx);
      Sf[tid * 128 + j] = e; sum += e;
    }
    float inv = 1.0f / sum;
    for (int j = 0; j < 128; ++j) Sf[tid * 128 + j] *= inv;
  }
  __syncthreads();

  // O = P @ V : 64x32 = 4x2 tiles; wave w does tile w; K=128 (4 WMMAs)
  {
    int tm = wave >> 1, tn = wave & 1;
    v8f acc = {};
    for (int kb = 0; kb < 128; kb += 32) {
      v16h a, b;
#pragma unroll
      for (int hh = 0; hh < 16; ++hh) {
        int ka = kb + hh + 16 * g;
        a[hh] = (_Float16)Sf[(tm * 16 + idx) * 128 + ka];
        int kw = kb + hh + 16 * g;
        b[hh] = Vh[kw * DH + tn * 16 + idx];
      }
      acc = __builtin_amdgcn_wmma_f32_16x16x32_f16(false, a, false, b,
                                                   (short)0, acc, false, false);
    }
#pragma unroll
    for (int r = 0; r < 8; ++r) {
      int qb = qc * 64 + tm * 16 + r + 8 * g;
      int d  = tn * 16 + idx;
      O[((size_t)qb * LL + l) * CC + h * DH + d] = acc[r];
    }
  }
}

// =====================================================================
// out[b,c] = mean_l mo[b,l,c]
// =====================================================================
__global__ __launch_bounds__(128) void k_mean(const float* __restrict__ mo,
                                              float* __restrict__ out)
{
  int b = blockIdx.x, c = threadIdx.x;
  float s = 0.0f;
  for (int l = 0; l < LL; ++l) s += mo[((size_t)b * LL + l) * CC + c];
  out[(size_t)b * CC + c] = s / (float)LL;
}

// =====================================================================
// host side
// =====================================================================
static void gemm(hipStream_t s, const float* A, int lda, const float* Wt, int ldw,
                 const float* bias, float* Out, int ldo, int M, int N, int K,
                 int ep, bool al16)
{
  dim3 g((N + 15) / 16, (M + 15) / 16);
  if (al16)
    hipLaunchKernelGGL((k_gemm_wmma<true>), g, dim3(32), 0, s,
                       A, lda, Wt, ldw, bias, Out, ldo, M, N, K, ep);
  else
    hipLaunchKernelGGL((k_gemm_wmma<false>), g, dim3(32), 0, s,
                       A, lda, Wt, ldw, bias, Out, ldo, M, N, K, ep);
}

extern "C" void kernel_launch(void* const* d_in, const int* in_sizes, int n_in,
                              void* d_out, int out_size, void* d_ws, size_t ws_size,
                              hipStream_t stream)
{
  (void)in_sizes; (void)n_in; (void)out_size; (void)ws_size;
  const float* x        = (const float*)d_in[0];
  // m1
  const float* m1_inw   = (const float*)d_in[1];   // (512,128)
  const float* m1_cw    = (const float*)d_in[2];   // (256,4)
  const float* m1_cb    = (const float*)d_in[3];
  const float* m1_xpw   = (const float*)d_in[4];   // (72,256)
  const float* m1_dtw   = (const float*)d_in[5];   // (256,8)
  const float* m1_dtb   = (const float*)d_in[6];
  const float* m1_Alog  = (const float*)d_in[7];   // (256,32)
  const float* m1_D     = (const float*)d_in[8];
  const float* m1_ow    = (const float*)d_in[9];   // (128,256)
  // m2
  const float* m2_inw   = (const float*)d_in[10];  // (324,81)
  const float* m2_cw    = (const float*)d_in[11];
  const float* m2_cb    = (const float*)d_in[12];
  const float* m2_xpw   = (const float*)d_in[13];  // (70,162)
  const float* m2_dtw   = (const float*)d_in[14];  // (162,6)
  const float* m2_dtb   = (const float*)d_in[15];
  const float* m2_Alog  = (const float*)d_in[16];  // (162,32)
  const float* m2_D     = (const float*)d_in[17];
  const float* m2_ow    = (const float*)d_in[18];  // (81,162)
  // attention
  const float* aA_inw   = (const float*)d_in[19];  // (384,128)
  const float* aA_inb   = (const float*)d_in[20];
  const float* aA_ow    = (const float*)d_in[21];  // (128,128)
  const float* aA_ob    = (const float*)d_in[22];
  const float* aB_inw   = (const float*)d_in[23];
  const float* aB_inb   = (const float*)d_in[24];
  const float* aB_ow    = (const float*)d_in[25];
  const float* aB_ob    = (const float*)d_in[26];
  // layernorms
  const float* lnpre_g  = (const float*)d_in[27];
  const float* lnpre_b  = (const float*)d_in[28];
  const float* lnpost_g = (const float*)d_in[29];
  const float* lnpost_b = (const float*)d_in[30];
  const float* lnspa_g  = (const float*)d_in[31];
  const float* lnspa_b  = (const float*)d_in[32];
  const float* lnspe_g  = (const float*)d_in[33];
  const float* lnspe_b  = (const float*)d_in[34];

  float* W = (float*)d_ws;
  const size_t NT = (size_t)T1 * CC;   // 1,327,104
  // persistent
  float* seq  = W;
  float* spa  = seq  + NT;
  float* spe  = spa  + NT;
  float* aspa = spe  + NT;
  float* aspe = aspa + NT;
  float* P0   = aspe + NT;             // reusable pool

  // ---------- stage 0: transpose + ln_pre ----------
  hipLaunchKernelGGL(k_seq_ln, dim3(T1), dim3(128), 0, stream,
                     x, lnpre_g, lnpre_b, seq);

  // ---------- mamba 1 (spatial, T=81, di=256) ----------
  {
    float* xz1   = P0;                            // T1*512
    float* u1    = xz1   + (size_t)T1 * 512;      // T1*256
    float* xdbl1 = u1    + (size_t)T1 * DI1;      // T1*72
    float* dt1   = xdbl1 + (size_t)T1 * W1n;      // T1*256
    float* ym1   = dt1   + (size_t)T1 * DI1;      // T1*256

    gemm(stream, seq, CC, m1_inw, CC, nullptr, xz1, 2 * DI1, T1, 2 * DI1, CC, 0, true);
    {
      size_t tot = (size_t)BN * LL * DI1;
      hipLaunchKernelGGL(k_conv_silu, dim3((unsigned)((tot + 255) / 256)), dim3(256),
                         0, stream, xz1, m1_cw, m1_cb, u1, BN, LL, DI1);
    }
    gemm(stream, u1, DI1, m1_xpw, DI1, nullptr, xdbl1, W1n, T1, W1n, DI1, 0, true);
    gemm(stream, xdbl1, W1n, m1_dtw, R1n, m1_dtb, dt1, DI1, T1, DI1, R1n, 2, false);
    hipLaunchKernelGGL(k_scan, dim3(BN), dim3(DI1), 0, stream,
                       u1, dt1, xdbl1, W1n, R1n, m1_Alog, m1_D, xz1, ym1, LL, DI1);
    gemm(stream, ym1, DI1, m1_ow, DI1, nullptr, spa, CC, T1, CC, DI1, 0, true);
  }

  // ---------- mamba 2 (spectral, T=128 channel-tokens, di=162) ----------
  {
    float* seqT  = P0;                            // T2*81
    float* xz2   = seqT  + (size_t)T2 * LL;       // T2*324
    float* u2    = xz2   + (size_t)T2 * 2 * DI2;  // T2*162
    float* xdbl2 = u2    + (size_t)T2 * DI2;      // T2*70
    float* dt2   = xdbl2 + (size_t)T2 * W2n;      // T2*162
    float* ym2   = dt2   + (size_t)T2 * DI2;      // T2*162
    float* out2  = ym2   + (size_t)T2 * DI2;      // T2*81

    {
      size_t tot = (size_t)BN * LL * CC;
      hipLaunchKernelGGL(k_t_blc2bcl, dim3((unsigned)((tot + 255) / 256)), dim3(256),
                         0, stream, seq, seqT);
    }
    gemm(stream, seqT, LL, m2_inw, LL, nullptr, xz2, 2 * DI2, T2, 2 * DI2, LL, 0, false);
    {
      size_t tot = (size_t)BN * CC * DI2;
      hipLaunchKernelGGL(k_conv_silu, dim3((unsigned)((tot + 255) / 256)), dim3(256),
                         0, stream, xz2, m2_cw, m2_cb, u2, BN, CC, DI2);
    }
    gemm(stream, u2, DI2, m2_xpw, DI2, nullptr, xdbl2, W2n, T2, W2n, DI2, 0, false);
    gemm(stream, xdbl2, W2n, m2_dtw, R2n, m2_dtb, dt2, DI2, T2, DI2, R2n, 2, false);
    hipLaunchKernelGGL(k_scan, dim3(BN), dim3(192), 0, stream,
                       u2, dt2, xdbl2, W2n, R2n, m2_Alog, m2_D, xz2, ym2, CC, DI2);
    gemm(stream, ym2, DI2, m2_ow, DI2, nullptr, out2, LL, T2, LL, DI2, 0, false);
    {
      size_t tot = (size_t)BN * CC * LL;
      hipLaunchKernelGGL(k_t_bcl2blc, dim3((unsigned)((tot + 255) / 256)), dim3(256),
                         0, stream, out2, spe);
    }
  }

  // ---------- attention s2sp: q = spa, kv = spe ----------
  {
    float* qa = P0;
    float* ka = qa + NT;
    float* va = ka + NT;
    float* oa = va + NT;
    gemm(stream, spa, CC, aA_inw,           CC, aA_inb,       qa, CC, T1, CC, CC, 1, true);
    gemm(stream, spe, CC, aA_inw + 128*128, CC, aA_inb + 128, ka, CC, T1, CC, CC, 1, true);
    gemm(stream, spe, CC, aA_inw + 256*128, CC, aA_inb + 256, va, CC, T1, CC, CC, 1, true);
    hipLaunchKernelGGL(k_attn, dim3(LL * HEADS, 2), dim3(256), 0, stream, qa, ka, va, oa);
    gemm(stream, oa, CC, aA_ow, CC, aA_ob, aspa, CC, T1, CC, CC, 1, true);
  }
  // ---------- attention sp2s: q = spe, kv = spa ----------
  {
    float* qb = P0;
    float* kb = qb + NT;
    float* vb = kb + NT;
    float* ob = vb + NT;
    gemm(stream, spe, CC, aB_inw,           CC, aB_inb,       qb, CC, T1, CC, CC, 1, true);
    gemm(stream, spa, CC, aB_inw + 128*128, CC, aB_inb + 128, kb, CC, T1, CC, CC, 1, true);
    gemm(stream, spa, CC, aB_inw + 256*128, CC, aB_inb + 256, vb, CC, T1, CC, CC, 1, true);
    hipLaunchKernelGGL(k_attn, dim3(LL * HEADS, 2), dim3(256), 0, stream, qb, kb, vb, ob);
    gemm(stream, ob, CC, aB_ow, CC, aB_ob, aspe, CC, T1, CC, CC, 1, true);
  }

  // ---------- residuals + LNs + mean ----------
  {
    float* so = P0;            // spatial_out
    float* sp = so + NT;       // spectral_out
    float* mo = sp + NT;
    hipLaunchKernelGGL(k_ln3, dim3(T1), dim3(128), 0, stream,
                       spa, aspa, (const float*)nullptr, lnspa_g, lnspa_b, so);
    hipLaunchKernelGGL(k_ln3, dim3(T1), dim3(128), 0, stream,
                       spe, aspe, (const float*)nullptr, lnspe_g, lnspe_b, sp);
    hipLaunchKernelGGL(k_ln3, dim3(T1), dim3(128), 0, stream,
                       so, sp, seq, lnpost_g, lnpost_b, mo);
    hipLaunchKernelGGL(k_mean, dim3(BN), dim3(128), 0, stream, mo, (float*)d_out);
  }
}
